// JaccardIndex_25640954757369
// MI455X (gfx1250) — compile-verified, matrix operand order checked by reading, not run on hardware
//
#include <hip/hip_runtime.h>

// ---------------------------------------------------------------------------
// Binary Jaccard index, MI455X (gfx1250, wave32).
// HBM-bound streaming reduction: ~201 MB once-through -> ~8.6 us at 23.3 TB/s.
//  - global_load_b128 non-temporal loads (data > 192MB L2, touched once)
//  - per-thread integer popcounts, 64 elems/thread
//  - wave reduction via v_wmma_f32_16x16x32_f16 with B = ones (row-sum trick):
//      D[m][n] = sum_k A[m][k];  lanes L and L+16 share row M=L%16, so the
//      row sum is count(L)+count(L+16). Summing the 8 D VGPRs gives lanes
//      0-15 rows 0-7 and lanes 16-31 rows 8-15; one shfl_xor(16) completes
//      the 32-lane sum. Counts <= 64 -> exact in f16; accumulation in f32.
//  - LDS float atomics across 8 waves, one global f32 atomic per (plane,chunk)
//  - finalize kernel: IoU per plane, mean over 96 planes.
// ---------------------------------------------------------------------------

typedef __attribute__((ext_vector_type(4)))  float    v4f;
typedef __attribute__((ext_vector_type(16))) _Float16 v16h;
typedef __attribute__((ext_vector_type(8)))  float    v8f;

#define THRESH            0.5f
#define PLANES            96                      // B*C = 32*3
#define HW                (512 * 512)             // 262144 elems / plane
#define CHUNKS_PER_PLANE  16
#define THREADS           256                     // 8 wave32s
#define ELEMS_PER_CHUNK   (HW / CHUNKS_PER_PLANE) // 16384
#define ITERS             (ELEMS_PER_CHUNK / (THREADS * 4)) // 16 -> 64 elem/thr

// Wave32-wide integer sum via one WMMA (B = all-ones) + one shfl_xor.
__device__ __forceinline__ float wave_sum_wmma(int count) {
  v16h a = {};                       // zero A; only slot 0 carries this lane's count
  a[0] = (_Float16)count;            // counts <= 64 -> exact in f16
  v16h ones;
#pragma unroll
  for (int i = 0; i < 16; ++i) ones[i] = (_Float16)1.0f;
  v8f c = {};
  // D = A x Ones + 0 : every D row m = count(lane m) + count(lane m+16)
  v8f d = __builtin_amdgcn_wmma_f32_16x16x32_f16(
      /*neg_a=*/false, a, /*neg_b=*/false, ones,
      /*c_mod=*/(short)0, c, /*reuse_a=*/false, /*reuse_b=*/false);
  float s = d[0] + d[1] + d[2] + d[3] + d[4] + d[5] + d[6] + d[7];
  s += __shfl_xor(s, 16, 32);        // combine row groups 0-7 and 8-15
  return s;                          // full 32-lane sum, in every lane
}

__global__ void jaccard_zero_ws(float* __restrict__ ws) {
  int t = threadIdx.x;
  if (t < PLANES * 3) ws[t] = 0.0f;
}

__global__ void __launch_bounds__(THREADS)
jaccard_partial(const float* __restrict__ in, const float* __restrict__ tg,
                float* __restrict__ ws) {
  const int plane   = blockIdx.x;
  const int chunk   = blockIdx.y;
  const int t       = threadIdx.x;
  const size_t base = (size_t)plane * HW + (size_t)chunk * ELEMS_PER_CHUNK;

  int ci = 0, ca = 0, cb = 0;
#pragma unroll
  for (int it = 0; it < ITERS; ++it) {
    const size_t idx = base + (size_t)it * (THREADS * 4) + (size_t)t * 4;
    v4f x = __builtin_nontemporal_load((const v4f*)(in + idx)); // global_load_b128 NT
    v4f y = __builtin_nontemporal_load((const v4f*)(tg + idx));
#pragma unroll
    for (int j = 0; j < 4; ++j) {
      int av = (x[j] >= THRESH);
      int bv = (y[j] >= THRESH);
      ca += av;
      cb += bv;
      ci += av & bv;
    }
  }

  // WMMA wave reductions (EXEC all-ones here: no divergence up to this point).
  float wi = wave_sum_wmma(ci);
  float wa = wave_sum_wmma(ca);
  float wb = wave_sum_wmma(cb);

  // Cross-wave combine in LDS (ds_add_f32), then one global atomic per block.
  __shared__ float sh[3];
  if (t < 3) sh[t] = 0.0f;
  __syncthreads();
  if ((t & 31) == 0) {
    atomicAdd(&sh[0], wi);
    atomicAdd(&sh[1], wa);
    atomicAdd(&sh[2], wb);
  }
  __syncthreads();
  if (t == 0) {
    atomicAdd(&ws[plane * 3 + 0], sh[0]);  // global_atomic_add_f32
    atomicAdd(&ws[plane * 3 + 1], sh[1]);
    atomicAdd(&ws[plane * 3 + 2], sh[2]);
  }
}

__global__ void jaccard_finalize(const float* __restrict__ ws,
                                 float* __restrict__ out) {
  __shared__ float sh[128];
  const int t = threadIdx.x;
  float iou = 0.0f;
  if (t < PLANES) {
    const float inter = ws[3 * t + 0];
    const float sa    = ws[3 * t + 1];
    const float sb    = ws[3 * t + 2];
    const float uni   = sa + sb - inter;
    iou = (uni > 0.0f) ? (inter / uni) : 1.0f;
  }
  sh[t] = iou;
  __syncthreads();
#pragma unroll
  for (int s = 64; s > 0; s >>= 1) {
    if (t < s) sh[t] += sh[t + s];
    __syncthreads();
  }
  if (t == 0) *out = sh[0] * (1.0f / (float)PLANES);
}

extern "C" void kernel_launch(void* const* d_in, const int* in_sizes, int n_in,
                              void* d_out, int out_size, void* d_ws, size_t ws_size,
                              hipStream_t stream) {
  (void)in_sizes; (void)n_in; (void)out_size; (void)ws_size;
  const float* in  = (const float*)d_in[0];
  const float* tg  = (const float*)d_in[1];
  float*       ws  = (float*)d_ws;       // 96 planes x {inter, cntA, cntB}
  float*       out = (float*)d_out;

  jaccard_zero_ws<<<1, 512, 0, stream>>>(ws);
  jaccard_partial<<<dim3(PLANES, CHUNKS_PER_PLANE), THREADS, 0, stream>>>(in, tg, ws);
  jaccard_finalize<<<1, 128, 0, stream>>>(ws, out);
}
